// ColumnSelfAttention_12506944766492
// MI455X (gfx1250) — matrix operand order repaired and checked by reading, not compiled, not run on hardware
//
#include <hip/hip_runtime.h>
#include <hip/hip_bf16.h>
#include <cstdint>
#include <cstddef>

// ---------------- problem constants ----------------
#define RR  128   // rows (attention axis)
#define CCn 256   // cols
#define EE  768   // embed
#define HHn 12    // heads
#define DKK 64    // head dim
#define MM  (RR * CCn)          // 32768 GEMM rows
#define LDP 72    // padded LDS row stride (elems): 36 dwords -> conflict-free
// ---------------------------------------------------

typedef __bf16 v4bf  __attribute__((ext_vector_type(4)));
typedef __bf16 v8bf  __attribute__((ext_vector_type(8)));
typedef __bf16 v16bf __attribute__((ext_vector_type(16)));
typedef float  v8f   __attribute__((ext_vector_type(8)));

// scheduler pinning: group `size` instrs of class `mask` (0x100=DS-read, 0x8=WMMA)
#if __has_builtin(__builtin_amdgcn_sched_group_barrier)
#define SCHED_GROUP(mask, size, id) __builtin_amdgcn_sched_group_barrier(mask, size, id)
#else
#define SCHED_GROUP(mask, size, id)
#endif

static __device__ __forceinline__ v8f wmma_bf16(v16bf a, v16bf b, v8f c) {
  // D = A(16x32 bf16) x B(32x16 bf16) + C(16x16 f32)
  return __builtin_amdgcn_wmma_f32_16x16x32_bf16(
      /*neg_a=*/false, a, /*neg_b=*/false, b,
      /*c_mod=*/(short)0, c, /*reuse_a=*/false, /*reuse_b=*/false);
}

static __device__ __forceinline__ v8f vzero8() {
  v8f z;
#pragma unroll
  for (int i = 0; i < 8; ++i) z[i] = 0.0f;
  return z;
}

// ---- CDNA5 async copy: global -> LDS, 16B per lane, tracked by ASYNCcnt ----
static __device__ __forceinline__ void async_load_b128(void* lds, const void* gbase,
                                                       uint32_t goff) {
  uint32_t l = (uint32_t)(uintptr_t)lds;  // LDS aperture: offset lives in addr[31:0]
  asm volatile("global_load_async_to_lds_b128 %0, %1, %2"
               :: "v"(l), "v"(goff), "s"(gbase)
               : "memory");
}
static __device__ __forceinline__ void wait_asynccnt_0() {
  asm volatile("s_wait_asynccnt 0x0" ::: "memory");
}
static __device__ __forceinline__ void wait_asynccnt_6() {
  asm volatile("s_wait_asynccnt 0x6" ::: "memory");
}

// A-fragment (16x32 bf16, M x K): lane m = lane&15, half = lane>>4.
// caller passes p = rowptr(m) + kbase + half*8 ; elems [p..p+7],[p+16..p+23]
static __device__ __forceinline__ v16bf load_fragA(const __bf16* p) {
  v8bf lo = *reinterpret_cast<const v8bf*>(p);
  v8bf hi = *reinterpret_cast<const v8bf*>(p + 16);
  return __builtin_shufflevector(lo, hi, 0,1,2,3,4,5,6,7,8,9,10,11,12,13,14,15);
}

// B-fragment (32x16 bf16, K x N) stored "N-major": lane n = lane&15;
// caller passes p = colptr(n) + kbase + half*8 ; elems [p..p+7],[p+16..p+23]
static __device__ __forceinline__ v16bf load_fragB(const __bf16* p) {
  v8bf lo = *reinterpret_cast<const v8bf*>(p);
  v8bf hi = *reinterpret_cast<const v8bf*>(p + 16);
  return __builtin_shufflevector(lo, hi, 0,1,2,3,4,5,6,7,8,9,10,11,12,13,14,15);
}

// ---------------- f32 -> bf16 conversion ----------------
__global__ void cvt_f32_to_bf16(const float* __restrict__ in,
                                __bf16* __restrict__ out, int n) {
  int i = (blockIdx.x * blockDim.x + threadIdx.x) * 4;
  if (i + 3 < n) {
    float4 f = *reinterpret_cast<const float4*>(in + i);
    v4bf o;
    o[0] = (__bf16)f.x; o[1] = (__bf16)f.y; o[2] = (__bf16)f.z; o[3] = (__bf16)f.w;
    *reinterpret_cast<v4bf*>(out + i) = o;
  } else {
    for (; i < n; ++i) out[i] = (__bf16)in[i];
  }
}

// ---------------- GEMM: Y[m][n] = (sum_k A[m][k]*W[n][k] + bias[n]) * scale ----
// Double-buffered async-LDS pipeline. Block = 256 thr (8 waves).
// Block tile: 128(M) x 64(N), K staged 64 at a time.
// Compute: pin 20 DS reads, then burst 8 back-to-back WMMAs (one dscnt wait).
__global__ void gemm_xwT(const __bf16* __restrict__ A, const __bf16* __restrict__ W,
                         const float* __restrict__ bias, void* __restrict__ out,
                         int out_f32, float scale, int M, int N, int K) {
  __shared__ __align__(16) __bf16 sA[2][128 * LDP];
  __shared__ __align__(16) __bf16 sB[2][64 * LDP];

  const int tid  = threadIdx.x;
  const int lane = tid & 31;
  const int wv   = tid >> 5;
  const int ln   = lane & 15;
  const int half = lane >> 4;
  const int row0 = blockIdx.x * 128;
  const int col0 = blockIdx.y * 64;

  auto stage = [&](int kt, int buf) {
#pragma unroll
    for (int it = 0; it < 4; ++it) {                 // A tile: 128x64 = 1024 x 16B
      const int idx = tid + it * 256;
      const int r   = idx >> 3;
      const int ko  = (idx & 7) * 8;
      async_load_b128(&sA[buf][r * LDP + ko], A,
                      (uint32_t)(((size_t)(row0 + r) * K + kt + ko) * 2));
    }
#pragma unroll
    for (int it = 0; it < 2; ++it) {                 // B tile: 64x64 = 512 x 16B
      const int idx = tid + it * 256;
      const int nr  = idx >> 3;
      const int ko  = (idx & 7) * 8;
      async_load_b128(&sB[buf][nr * LDP + ko], W,
                      (uint32_t)(((size_t)(col0 + nr) * K + kt + ko) * 2));
    }
  };

  v8f acc[4];
#pragma unroll
  for (int cb = 0; cb < 4; ++cb) acc[cb] = vzero8();

  const int NS = K / 64;                             // 12 stages
  stage(0, 0);
  for (int s = 0; s < NS; ++s) {
    const int buf = s & 1;
    if (s + 1 < NS) { stage((s + 1) * 64, buf ^ 1); wait_asynccnt_6(); }
    else            { wait_asynccnt_0(); }
    __syncthreads();                                 // publish stage s to all waves

    // ---- batch all fragment loads, then burst the WMMAs ----
    v16bf af[2];
#pragma unroll
    for (int x = 0; x < 2; ++x)
      af[x] = load_fragA(&sA[buf][(wv * 16 + ln) * LDP + x * 32 + half * 8]);
    v16bf bf[2][4];
#pragma unroll
    for (int x = 0; x < 2; ++x)
#pragma unroll
      for (int cb = 0; cb < 4; ++cb)
        bf[x][cb] = load_fragB(&sB[buf][(cb * 16 + ln) * LDP + x * 32 + half * 8]);
#pragma unroll
    for (int x = 0; x < 2; ++x)
#pragma unroll
      for (int cb = 0; cb < 4; ++cb)
        acc[cb] = wmma_bf16(af[x], bf[x][cb], acc[cb]);

    SCHED_GROUP(0x100, 20, 0);   // all 20 ds_load_b128 first
    SCHED_GROUP(0x008,  8, 0);   // then 8 back-to-back WMMAs

    __syncthreads();                                 // buffer safe to overwrite
  }

#pragma unroll
  for (int cb = 0; cb < 4; ++cb) {
    const int col = col0 + cb * 16 + ln;
    const float b = bias[col];
#pragma unroll
    for (int e = 0; e < 8; ++e) {
      const int row = row0 + wv * 16 + half * 8 + e; // C/D: VGPR e <-> M = e + 8*half
      const float y = (acc[cb][e] + b) * scale;
      if (out_f32) ((float*)out)[(size_t)row * N + col] = y;
      else         ((__bf16*)out)[(size_t)row * N + col] = (__bf16)y;
    }
  }
}

// ---------------- fused attention per (h, c) ----------------
// q,k,v: (MM x EE) bf16 (row index = r*CCn + c, col = h*64 + d); q pre-scaled.
// probs: (H, C, R, R) f32 out.  ctx: (MM x EE) bf16 out.
// Block = 128 threads (4 waves); wave w owns query rows [32w, 32w+32).
__global__ void attn_kernel(const __bf16* __restrict__ q, const __bf16* __restrict__ k,
                            const __bf16* __restrict__ v,
                            const unsigned char* __restrict__ mask,
                            float* __restrict__ probs, __bf16* __restrict__ ctx) {
  __shared__ __align__(16) __bf16 Ks[RR * LDP];     // K tile, shared by all 4 waves
  __shared__ __align__(16) __bf16 Vt[DKK * 136];    // V transposed, padded stride
  __shared__ __align__(16) __bf16 Ps[RR * 136];     // probs in bf16

  const int h    = blockIdx.x >> 8;     // blockIdx.x = h*256 + c
  const int c    = blockIdx.x & 255;
  const int tid  = threadIdx.x;
  const int lane = tid & 31;
  const int wv   = tid >> 5;
  const int ln   = lane & 15;
  const int half = lane >> 4;

  // Async-stage K tile: 128 rows x 64 elems = 1024 x 16B chunks, 8 per thread
#pragma unroll
  for (int it = 0; it < 8; ++it) {
    const int idx = tid + it * 128;
    const int j   = idx >> 3;
    const int ko  = (idx & 7) * 8;
    async_load_b128(&Ks[j * LDP + ko], k,
                    (uint32_t)((((size_t)j * CCn + c) * EE + h * DKK + ko) * 2));
  }

  // Stage V transposed (scalar; it's a transpose): Vt[d][j] = V[j][d]
  for (int i = tid; i < RR * DKK; i += 128) {
    const int j = i >> 6, d = i & 63;
    Vt[d * 136 + j] = v[((size_t)j * CCn + c) * EE + h * DKK + d];
  }

  wait_asynccnt_0();
  __syncthreads();                       // Ks (+Vt) visible to all waves

  // ---- phase 1: S = Q K^T, mask, softmax, emit probs (f32 global + bf16 LDS) ----
  for (int rb2 = 0; rb2 < 2; ++rb2) {
    const int rb = wv * 2 + rb2;
    const __bf16* qrow = q + ((size_t)(rb * 16 + ln) * CCn + c) * EE + h * DKK;
    v16bf a0 = load_fragA(qrow + 0  + half * 8);   // K = 0..31
    v16bf a1 = load_fragA(qrow + 32 + half * 8);   // K = 32..63

    v8f acc[8];
#pragma unroll
    for (int cb = 0; cb < 8; ++cb) acc[cb] = vzero8();

    // two groups of 4 col-blocks: load 8 K-frags, then 8 back-to-back WMMAs
#pragma unroll
    for (int g = 0; g < 2; ++g) {
      v16bf bf[4][2];
#pragma unroll
      for (int c4 = 0; c4 < 4; ++c4) {
        const __bf16* krow = &Ks[((g * 4 + c4) * 16 + ln) * LDP];
        bf[c4][0] = load_fragB(krow + 0  + half * 8);
        bf[c4][1] = load_fragB(krow + 32 + half * 8);
      }
#pragma unroll
      for (int c4 = 0; c4 < 4; ++c4) {
        acc[g * 4 + c4] = wmma_bf16(a0, bf[c4][0], acc[g * 4 + c4]);
        acc[g * 4 + c4] = wmma_bf16(a1, bf[c4][1], acc[g * 4 + c4]);
      }
      SCHED_GROUP(0x100, 16, 0);   // 16 ds_load_b128 first
      SCHED_GROUP(0x008,  8, 0);   // then 8 WMMAs
    }

    // key padding mask: key j masked if padding_mask[0][j][c]
#pragma unroll
    for (int cb = 0; cb < 8; ++cb) {
      if (mask[(size_t)(cb * 16 + ln) * CCn + c]) {
#pragma unroll
        for (int e = 0; e < 8; ++e) acc[cb][e] = -10000.0f;
      }
    }

    // softmax over j: row (e,half) spans elem e of 8 accs across 16 lanes
#pragma unroll
    for (int e = 0; e < 8; ++e) {
      float mx = acc[0][e];
#pragma unroll
      for (int cb = 1; cb < 8; ++cb) mx = fmaxf(mx, acc[cb][e]);
      for (int off = 1; off < 16; off <<= 1) mx = fmaxf(mx, __shfl_xor(mx, off, 32));
      float sum = 0.0f;
#pragma unroll
      for (int cb = 0; cb < 8; ++cb) {
        float ex = __expf(acc[cb][e] - mx);
        acc[cb][e] = ex;
        sum += ex;
      }
      for (int off = 1; off < 16; off <<= 1) sum += __shfl_xor(sum, off, 32);
      const float inv = 1.0f / sum;

      const int i = rb * 16 + half * 8 + e;
      float* prow = probs + ((size_t)blockIdx.x * RR + i) * RR;
#pragma unroll
      for (int cb = 0; cb < 8; ++cb) {
        const float p = acc[cb][e] * inv;
        prow[cb * 16 + ln] = p;
        Ps[i * 136 + cb * 16 + ln] = (__bf16)p;
      }
    }
  }

  __syncthreads();

  // ---- phase 2: ctx = P V  (A from Ps LDS, B from Vt LDS) ----
  for (int rb2 = 0; rb2 < 2; ++rb2) {
    const int rb = wv * 2 + rb2;
    v8f cacc[4];
#pragma unroll
    for (int dt = 0; dt < 4; ++dt) cacc[dt] = vzero8();

#pragma unroll
    for (int kt = 0; kt < 4; ++kt) {   // K = 128 in steps of 32
      v16bf af = load_fragA(Ps + (rb * 16 + ln) * 136 + kt * 32 + half * 8);
      v16bf bf[4];
#pragma unroll
      for (int dt = 0; dt < 4; ++dt)
        bf[dt] = load_fragB(Vt + (dt * 16 + ln) * 136 + kt * 32 + half * 8);
#pragma unroll
      for (int dt = 0; dt < 4; ++dt)
        cacc[dt] = wmma_bf16(af, bf[dt], cacc[dt]);
      SCHED_GROUP(0x100, 10, 0);   // 10 ds_load_b128 first
      SCHED_GROUP(0x008,  4, 0);   // then 4 WMMAs
    }

#pragma unroll
    for (int dt = 0; dt < 4; ++dt) {
#pragma unroll
      for (int e = 0; e < 8; ++e) {
        const int i = rb * 16 + half * 8 + e;
        ctx[((size_t)i * CCn + c) * EE + h * DKK + dt * 16 + ln] = (__bf16)cacc[dt][e];
      }
    }
  }
}

// ---------------- host side ----------------
extern "C" void kernel_launch(void* const* d_in, const int* in_sizes, int n_in,
                              void* d_out, int out_size, void* d_ws, size_t ws_size,
                              hipStream_t stream) {
  (void)in_sizes; (void)n_in; (void)out_size; (void)ws_size;

  const float*         x    = (const float*)d_in[0];
  const unsigned char* mask = (const unsigned char*)d_in[1];  // numpy bool = 1 byte
  const float* Wq = (const float*)d_in[2]; const float* bq = (const float*)d_in[3];
  const float* Wk = (const float*)d_in[4]; const float* bk = (const float*)d_in[5];
  const float* Wv = (const float*)d_in[6]; const float* bv = (const float*)d_in[7];
  const float* Wo = (const float*)d_in[8]; const float* bo = (const float*)d_in[9];

  float* out   = (float*)d_out;                       // (R,C,B,E) = 25165824 f32
  float* probs = out + (size_t)MM * EE;               // (H,C,B,R,R) = 50331648 f32

  char* ws = (char*)d_ws;
  const size_t XB_BYTES = (size_t)MM * EE * 2;        // 50331648
  const size_t W_BYTES  = (size_t)EE * EE * 2;        // 1179648
  __bf16* xb   = (__bf16*)ws;
  __bf16* wqb  = (__bf16*)(ws + XB_BYTES);
  __bf16* wkb  = wqb + (size_t)EE * EE;
  __bf16* wvb  = wkb + (size_t)EE * EE;
  __bf16* wob  = wvb + (size_t)EE * EE;
  __bf16* qb   = (__bf16*)(ws + XB_BYTES + 4 * W_BYTES);
  __bf16* kb   = qb + (size_t)MM * EE;
  __bf16* vb   = kb + (size_t)MM * EE;
  __bf16* ctxb = xb;   // safe reuse: x is dead after the V projection completes

  const int NX = MM * EE;      // 25165824
  const int NW = EE * EE;      // 589824

  cvt_f32_to_bf16<<<dim3((NX / 4 + 255) / 256), dim3(256), 0, stream>>>(x,  xb,  NX);
  cvt_f32_to_bf16<<<dim3((NW / 4 + 255) / 256), dim3(256), 0, stream>>>(Wq, wqb, NW);
  cvt_f32_to_bf16<<<dim3((NW / 4 + 255) / 256), dim3(256), 0, stream>>>(Wk, wkb, NW);
  cvt_f32_to_bf16<<<dim3((NW / 4 + 255) / 256), dim3(256), 0, stream>>>(Wv, wvb, NW);
  cvt_f32_to_bf16<<<dim3((NW / 4 + 255) / 256), dim3(256), 0, stream>>>(Wo, wob, NW);

  const dim3 ggrid(MM / 128, EE / 64);   // 256 x 12
  const float qscale = 0.125f;           // dk^-0.5, dk = 64
  gemm_xwT<<<ggrid, dim3(256), 0, stream>>>(xb, wqb, bq, qb, 0, qscale, MM, EE, EE);
  gemm_xwT<<<ggrid, dim3(256), 0, stream>>>(xb, wkb, bk, kb, 0, 1.0f,   MM, EE, EE);
  gemm_xwT<<<ggrid, dim3(256), 0, stream>>>(xb, wvb, bv, vb, 0, 1.0f,   MM, EE, EE);

  attn_kernel<<<dim3(HHn * CCn), dim3(128), 0, stream>>>(qb, kb, vb, mask, probs, ctxb);

  gemm_xwT<<<ggrid, dim3(256), 0, stream>>>(ctxb, wob, bo, out, 1, 1.0f, MM, EE, EE);
}
